// AttentionHead_17179869184650
// MI455X (gfx1250) — matrix operand order repaired and checked by reading, not compiled, not run on hardware
//
#include <hip/hip_runtime.h>
#include <cstdint>

// Problem dims (from reference)
#define BATCH    512
#define SEQ      200
#define DMODEL   1024
#define DHEAD    64
#define TPAD     224          // 14 tiles of 16 (pad so 32-wide s-chunks stay in range)
#define NT_PROJ  14           // t-tiles written by projection (covers rows 0..223, zero-padded)
#define NT_ATTN  13           // t-tiles that contain real output rows

typedef __attribute__((ext_vector_type(16))) __bf16 v16bf;
typedef __attribute__((ext_vector_type(8)))  __bf16 v8bf;
typedef __attribute__((ext_vector_type(8)))  float  v8f;

// ---------------------------------------------------------------------------
// WMMA helper: D = A(16x32 bf16) * B(32x16 bf16) + C(16x16 f32)
// ---------------------------------------------------------------------------
__device__ __forceinline__ v8f wmma_bf16(v16bf a, v16bf b, v8f c) {
  return __builtin_amdgcn_wmma_f32_16x16x32_bf16(
      /*neg_a=*/false, a, /*neg_b=*/false, b,
      /*c_mod=*/(short)0, c, /*reuse_a=*/false, /*reuse_b=*/false);
}

// A-fragment element order (ISA 7.12.2, 16-bit A 16x32): per lane, elements
// 0..7 = K of [off..off+8), elements 8..15 = K of [off+16..off+24).
__device__ __forceinline__ v16bf ldA(const __bf16* rowp, int off) {
  v8bf lo = *(const v8bf*)(rowp + off);
  v8bf hi = *(const v8bf*)(rowp + off + 16);
  return __builtin_shufflevector(lo, hi, 0,1,2,3,4,5,6,7,8,9,10,11,12,13,14,15);
}

// B-fragment: p points at column N of the tile (K stride ld); elements
// 0..7 = K of [off..off+8), elements 8..15 = K of [off+16..off+24).
__device__ __forceinline__ v16bf ldB(const __bf16* p, int ld, int off) {
  v16bf r;
#pragma unroll
  for (int s = 0; s < 8; ++s) {
    r[s]     = p[(size_t)(off + s) * ld];
    r[8 + s] = p[(size_t)(off + 16 + s) * ld];
  }
  return r;
}

// 8x f32 -> v8bf
__device__ __forceinline__ v8bf cvt8(float4 a, float4 b) {
  v8bf r;
  r[0]=(__bf16)a.x; r[1]=(__bf16)a.y; r[2]=(__bf16)a.z; r[3]=(__bf16)a.w;
  r[4]=(__bf16)b.x; r[5]=(__bf16)b.y; r[6]=(__bf16)b.z; r[7]=(__bf16)b.w;
  return r;
}

// Butterfly reductions across the 16-lane half that shares a C-matrix row.
__device__ __forceinline__ float rmax16(float v) {
  v = fmaxf(v, __shfl_xor(v, 1, 32));
  v = fmaxf(v, __shfl_xor(v, 2, 32));
  v = fmaxf(v, __shfl_xor(v, 4, 32));
  v = fmaxf(v, __shfl_xor(v, 8, 32));
  return v;
}
__device__ __forceinline__ float rsum16(float v) {
  v += __shfl_xor(v, 1, 32);
  v += __shfl_xor(v, 2, 32);
  v += __shfl_xor(v, 4, 32);
  v += __shfl_xor(v, 8, 32);
  return v;
}

// ---------------------------------------------------------------------------
// Phase 0: one-shot weight f32 -> bf16 pre-conversion (W{q,k,v} -> ws).
// 3 x 1024x64; removes per-iteration weight conversions from the hot loop
// and halves weight L2 traffic for the 7168 projection blocks.
// ---------------------------------------------------------------------------
__global__ __launch_bounds__(256)
void wcvt_kernel(const float* __restrict__ Wq,
                 const float* __restrict__ Wk,
                 const float* __restrict__ Wv,
                 unsigned short* __restrict__ Wb_)
{
  const float* src = (blockIdx.y == 0) ? Wq : (blockIdx.y == 1) ? Wk : Wv;
  __bf16* dst = (__bf16*)Wb_ + (size_t)blockIdx.y * (DMODEL * DHEAD);
  const int base = (blockIdx.x * 256 + threadIdx.x) * 8;
  float4 a = *(const float4*)(src + base);
  float4 b = *(const float4*)(src + base + 4);
  *(v8bf*)(dst + base) = cvt8(a, b);
}

// ---------------------------------------------------------------------------
// Phase 1: fused QKV projection (bf16 out). One block = one (b, 16-row tile).
//  - q tile staged f32 into LDS via CDNA5 async global->LDS copies
//  - converted to bf16 ONCE cooperatively (shared by all 4 waves)
//  - wave w computes N-columns [16w,16w+16) of Q, K, V with bf16 WMMA
// Q is pre-scaled by 1/sqrt(1024); K stored transposed Kt[b][d][s];
// rows t >= 200 stored as exact zeros (pads TPAD=224).
// ---------------------------------------------------------------------------
__global__ __launch_bounds__(128)
void qkv_proj_kernel(const float* __restrict__ q,
                     const unsigned short* __restrict__ Wb_,
                     unsigned short* __restrict__ Qb_,
                     unsigned short* __restrict__ Kt_,
                     unsigned short* __restrict__ Vb_)
{
  __shared__ __align__(16) float  qtile[16][DMODEL];   // 64 KB (async staged)
  __shared__ __align__(16) __bf16 qbf[16][DMODEL];     // 32 KB (converted once)
  const __bf16* Wqb = (const __bf16*)Wb_;
  const __bf16* Wkb = Wqb + DMODEL * DHEAD;
  const __bf16* Wvb = Wkb + DMODEL * DHEAD;
  __bf16* Qb = (__bf16*)Qb_;
  __bf16* Kt = (__bf16*)Kt_;
  __bf16* Vb = (__bf16*)Vb_;

  const int tTile = blockIdx.x;
  const int b     = blockIdx.y;
  const int t0    = tTile * 16;
  const int tid   = threadIdx.x;

  // ---- stage 16x1024 f32 tile of q into LDS (async, ASYNCcnt tracked) ----
  {
    const unsigned int ldsBase = (unsigned int)(uintptr_t)(&qtile[0][0]);
    for (int sweep = 0; sweep < 32; ++sweep) {
      int idx = sweep * 512 + tid * 4;          // flat f32 index in tile
      int row = idx >> 10;
      int col = idx & (DMODEL - 1);
      int rg  = t0 + row; if (rg > SEQ - 1) rg = SEQ - 1;   // clamp pad rows
      unsigned int goff = (unsigned int)(((b * SEQ + rg) << 10) + col) * 4u;
      unsigned int loff = ldsBase + (unsigned int)idx * 4u;
      asm volatile("global_load_async_to_lds_b128 %0, %1, %2"
                   :: "v"(loff), "v"(goff), "s"(q) : "memory");
    }
    asm volatile("s_wait_asynccnt 0" ::: "memory");
  }
  __syncthreads();

  // ---- convert the staged tile to bf16 once (all waves cooperate) ----
  {
    const float* src = &qtile[0][0];
    __bf16* dst = &qbf[0][0];
    for (int base = tid * 8; base < 16 * DMODEL; base += 128 * 8) {
      float4 a = *(const float4*)(src + base);
      float4 c = *(const float4*)(src + base + 4);
      *(v8bf*)(dst + base) = cvt8(a, c);
    }
  }
  __syncthreads();

  const int lane = tid & 31;
  const int wave = tid >> 5;
  const int n0   = wave * 16;
  const int m    = lane & 15;
  const int hi   = lane >> 4;
  const int kb   = hi << 3;                     // 0 or 8

  v8f cq = {}; v8f ck = {}; v8f cv = {};
#pragma unroll 4
  for (int k0 = 0; k0 < DMODEL; k0 += 32) {
    v16bf af = ldA(&qbf[m][k0], kb);            // 2x ds_load_b128

    const int krow = k0;                        // column n0+m, K stride DHEAD
    const size_t woff = (size_t)krow * DHEAD + n0 + m;
    v16bf bq = ldB(Wqb + woff, DHEAD, kb);
    v16bf bk = ldB(Wkb + woff, DHEAD, kb);
    v16bf bv = ldB(Wvb + woff, DHEAD, kb);

    cq = wmma_bf16(af, bq, cq);
    ck = wmma_bf16(af, bk, ck);
    cv = wmma_bf16(af, bv, cv);
  }

  // C layout: element (M,N) with M = g + 8*hi, N = lane&15 (per VGPR g)
  const int nn = n0 + (lane & 15);
#pragma unroll
  for (int g = 0; g < 8; ++g) {
    const int t = t0 + g + 8 * hi;
    const bool valid = (t < SEQ);
    __bf16 zq = valid ? (__bf16)(cq[g] * 0.03125f) : (__bf16)0.0f; // fold 1/sqrt(1024)
    __bf16 zk = valid ? (__bf16)ck[g] : (__bf16)0.0f;
    __bf16 zv = valid ? (__bf16)cv[g] : (__bf16)0.0f;
    Qb[((size_t)b * TPAD + t) * DHEAD + nn] = zq;
    Vb[((size_t)b * TPAD + t) * DHEAD + nn] = zv;
    Kt[((size_t)b * DHEAD + nn) * TPAD + t] = zk;   // transposed for B-frags
  }
}

// ---------------------------------------------------------------------------
// Phase 2: causal flash attention. One wave (block=32) per (b, 16-row tile).
// Per 32-key chunk: 4 WMMAs for S = Q*K^T, masked online softmax with shfl
// butterflies, P re-shaped C-layout -> A-layout through a 1KB LDS tile,
// then 4 WMMAs for O += P*V.
// ---------------------------------------------------------------------------
__global__ __launch_bounds__(32)
void flash_attn_kernel(const unsigned short* __restrict__ Qb_,
                       const unsigned short* __restrict__ Kt_,
                       const unsigned short* __restrict__ Vb_,
                       float* __restrict__ out)
{
  __shared__ __align__(16) __bf16 Pbuf[16][32];
  const __bf16* Qb = (const __bf16*)Qb_;
  const __bf16* Kt = (const __bf16*)Kt_;
  const __bf16* Vb = (const __bf16*)Vb_;

  const int i    = blockIdx.x;            // t-tile
  const int b    = blockIdx.y;
  const int t0   = i * 16;
  const int lane = threadIdx.x & 31;
  const int m    = lane & 15;
  const int hi   = lane >> 4;
  const int kb   = hi << 3;

  // Q A-fragments (d = 0..63 -> two 16x32 frags), bf16 already scaled
  const __bf16* qrow = Qb + ((size_t)b * TPAD + t0 + m) * DHEAD;
  const v16bf aq0 = ldA(qrow, kb);
  const v16bf aq1 = ldA(qrow, 32 + kb);

  float mr[8], lr[8];
  v8f o0 = {}, o1 = {}, o2 = {}, o3 = {};
#pragma unroll
  for (int g = 0; g < 8; ++g) { mr[g] = -3.0e38f; lr[g] = 0.0f; }

  const __bf16* ktb = Kt + (size_t)b * DHEAD * TPAD;
  const int nCh = (i >> 1) + 1;                 // 32-key chunks up to diagonal
  for (int ch = 0; ch < nCh; ++ch) {
    const int s0 = ch * 32;

    // S = Q * K^T for two 16-key subtiles
    const __bf16* k0p = ktb + (size_t)(0)  * TPAD + s0 + m;  // d base 0, key s0+m
    const __bf16* k1p = ktb + (size_t)(32) * TPAD + s0 + m;  // d base 32, key s0+m
    v8f S0 = {}, S1 = {};
    S0 = wmma_bf16(aq0, ldB(k0p, TPAD, kb),      S0);
    S0 = wmma_bf16(aq1, ldB(k1p, TPAD, kb),      S0);
    S1 = wmma_bf16(aq0, ldB(k0p + 16, TPAD, kb), S1);
    S1 = wmma_bf16(aq1, ldB(k1p + 16, TPAD, kb), S1);

    // causal mask + online softmax (rows striped: M = g + 8*hi, N = lane&15)
    const int sc0 = s0 + (lane & 15);
    const int sc1 = sc0 + 16;
#pragma unroll
    for (int g = 0; g < 8; ++g) {
      const int t = t0 + g + 8 * hi;
      const float x0 = (sc0 <= t) ? S0[g] : -3.0e38f;
      const float x1 = (sc1 <= t) ? S1[g] : -3.0e38f;
      const float tm  = rmax16(fmaxf(x0, x1));
      const float nmv = fmaxf(mr[g], tm);
      const float rv  = __expf(mr[g] - nmv);
      const float p0  = __expf(x0 - nmv);
      const float p1  = __expf(x1 - nmv);
      Pbuf[g + 8 * hi][lane & 15]        = (__bf16)p0;
      Pbuf[g + 8 * hi][16 + (lane & 15)] = (__bf16)p1;
      lr[g] = lr[g] * rv + rsum16(p0 + p1);
      mr[g] = nmv;
      o0[g] *= rv; o1[g] *= rv; o2[g] *= rv; o3[g] *= rv;
    }

    // fence the C->A reshape through LDS (same-wave ds ordering)
    asm volatile("s_wait_dscnt 0" ::: "memory");
    const v16bf ap = ldA(&Pbuf[m][0], kb);      // P as 16x32 A-fragment

    // O += P * V   (V row-major: column d = m, K = s-row stride DHEAD)
    const __bf16* vrow = Vb + ((size_t)b * TPAD + s0) * DHEAD + m;
    o0 = wmma_bf16(ap, ldB(vrow, DHEAD, kb),      o0);
    o1 = wmma_bf16(ap, ldB(vrow + 16, DHEAD, kb), o1);
    o2 = wmma_bf16(ap, ldB(vrow + 32, DHEAD, kb), o2);
    o3 = wmma_bf16(ap, ldB(vrow + 48, DHEAD, kb), o3);
  }

  // epilogue: normalize and store fp32 output rows t < 200
  float* obase = out + (size_t)b * SEQ * DHEAD;
  const int nn = lane & 15;
#pragma unroll
  for (int g = 0; g < 8; ++g) {
    const int t = t0 + g + 8 * hi;
    if (t < SEQ) {
      const float inv = 1.0f / lr[g];
      float* p = obase + (size_t)t * DHEAD + nn;
      p[0]  = o0[g] * inv;
      p[16] = o1[g] * inv;
      p[32] = o2[g] * inv;
      p[48] = o3[g] * inv;
    }
  }
}

// ---------------------------------------------------------------------------
// Host launch. Workspace layout (bf16 elements):
//   Qb | Vb | Kt   each 512*224*64  (~14.7 MB each)
//   Wb (Wq|Wk|Wv bf16)  3*1024*64   (384 KB)
// Total ~44.4 MB, fully rewritten every call (deterministic).
// ---------------------------------------------------------------------------
extern "C" void kernel_launch(void* const* d_in, const int* in_sizes, int n_in,
                              void* d_out, int out_size, void* d_ws, size_t ws_size,
                              hipStream_t stream)
{
  (void)in_sizes; (void)n_in; (void)out_size; (void)ws_size;
  const float* q  = (const float*)d_in[0];
  const float* Wq = (const float*)d_in[1];
  const float* Wk = (const float*)d_in[2];
  const float* Wv = (const float*)d_in[3];
  float* out = (float*)d_out;

  const size_t matElems = (size_t)BATCH * TPAD * DHEAD;
  unsigned short* Qb = (unsigned short*)d_ws;
  unsigned short* Vb = Qb + matElems;
  unsigned short* Kt = Vb + matElems;
  unsigned short* Wb = Kt + matElems;

  wcvt_kernel<<<dim3(DMODEL * DHEAD / (256 * 8), 3), dim3(256), 0, stream>>>(
      Wq, Wk, Wv, Wb);
  qkv_proj_kernel<<<dim3(NT_PROJ, BATCH), dim3(128), 0, stream>>>(
      q, Wb, Qb, Kt, Vb);
  flash_attn_kernel<<<dim3(NT_ATTN, BATCH), dim3(32), 0, stream>>>(
      Qb, Kt, Vb, out);
}
